// LocalTriangleAttentionNew_84731114815925
// MI455X (gfx1250) — matrix operand order repaired and checked by reading, not compiled, not run on hardware
//
#include <hip/hip_runtime.h>

// ---------------------------------------------------------------------------
// LocalTriangleAttention for MI455X (gfx1250, wave32, WMMA 16x16x32 bf16)
// B=1, N=384, C_Z=128, C_S=256, C_RBF=256, C_GATE=16, C_MUL=128, H=4, C_HID=32
// All GEMM-shaped math runs through v_wmma_f32_16x16x32_bf16 with f32 accum.
// A-tiles are staged to LDS with GLOBAL_LOAD_ASYNC_TO_LDS_B128 (ASYNCcnt).
// ---------------------------------------------------------------------------

#define NTOK 384
#define NN   (384 * 384)   // 147456
#define CZ   128
#define KTOT 32
#define MROWS (NTOK * KTOT) // 12288
#define INFV 1e9f

typedef unsigned short u16;
typedef __attribute__((ext_vector_type(16))) __bf16 bf16x16;
typedef __attribute__((ext_vector_type(8)))  float  f32x8;

union Frag { bf16x16 v; unsigned int d[8]; };

__device__ __forceinline__ u16 f2bf(float f) {
  union { float f; unsigned int u; } cv; cv.f = f;
  unsigned int u = cv.u;
  unsigned int r = u + 0x7FFFu + ((u >> 16) & 1u);   // round-to-nearest-even
  return (u16)(r >> 16);
}
__device__ __forceinline__ float sigm(float x) { return 1.f / (1.f + __expf(-x)); }

// K-offset pattern for the 16-bit A fragment (16x32): lanes 0-15 rows, halves
// select K groups; VGPR v -> K = 16*half + 2v (contiguous, matches B)
__device__ __forceinline__ int a_koff(int v, int half) {
  return 16 * half + 2 * v;
}

// wave32 butterfly sum
__device__ __forceinline__ float wave_sum(float s) {
#pragma unroll
  for (int m = 16; m >= 1; m >>= 1) s += __shfl_xor(s, m, 32);
  return s;
}

// async global -> LDS copy of 16 bytes (tracked by ASYNCcnt)
__device__ __forceinline__ void async_copy_b128(unsigned int lds_off, const void* gsrc) {
  asm volatile("global_load_async_to_lds_b128 %0, %1, off"
               :: "v"(lds_off), "v"(gsrc) : "memory");
}
__device__ __forceinline__ void async_fence() {
  asm volatile("s_wait_asynccnt 0x0" ::: "memory");
}

// ---------------------------------------------------------------------------
// Weight convert+transpose: W[K,Nc] f32 -> Wt[Nc,K] bf16 (B-fragment friendly)
// ---------------------------------------------------------------------------
__global__ void transpose_bf16_k(const float* __restrict__ W, u16* __restrict__ Wt,
                                 int K, int Nc) {
  int id = blockIdx.x * 256 + threadIdx.x;
  if (id >= K * Nc) return;
  int n = id % Nc, k = id / Nc;
  Wt[(size_t)n * K + k] = f2bf(W[id]);
}

// ---------------------------------------------------------------------------
// Generic WMMA GEMM: C[M,128] f32 = A[M,K] bf16 @ Wt[128,K] bf16 (+bias)
// Block = 256 threads (8 waves). Wave w owns output columns [16w,16w+16).
// Whole A row-block (16 x K, K<=256) async-copied to LDS once; single barrier;
// inner loop is a pure ds_load/global_load/wmma stream.
// ---------------------------------------------------------------------------
__global__ __launch_bounds__(256)
void gemm_bf16_k(const u16* __restrict__ A, const u16* __restrict__ Wt,
                 const float* __restrict__ bias, float* __restrict__ C,
                 int M, int K) {
  __shared__ u16 aTile[16 * 256];            // supports K up to 256
  const int tid = threadIdx.x;
  const int wave = tid >> 5, lane = tid & 31;
  const int m0 = blockIdx.x * 16;
  const int col0 = wave * 16;
  const int mrow = lane & 15, half = lane >> 4;

  // ---- async stage of the full 16xK A tile into LDS ----
  {
    const int cprShift = (K == 256) ? 5 : 4;          // 16B chunks per row
    const int cprMask = (1 << cprShift) - 1;
    const int nchunk = 2 * K;                         // (16*K*2)/16
    unsigned int ldsbase = (unsigned int)(__SIZE_TYPE__)aTile;
    for (int c = tid; c < nchunk; c += 256) {
      int r = c >> cprShift;
      int cc = c & cprMask;                           // 16B chunk within row
      const void* src = (const void*)(A + (size_t)(m0 + r) * K + cc * 8);
      async_copy_b128(ldsbase + (unsigned int)(c * 16), src);
    }
    async_fence();
  }
  __syncthreads();

  const int ldw = K >> 1;                             // LDS row stride in dwords
  const unsigned int* lds32 = (const unsigned int*)aTile;
  f32x8 acc = {};
  for (int k0 = 0; k0 < K; k0 += 32) {
    Frag af, bf;
#pragma unroll
    for (int v = 0; v < 8; ++v)
      af.d[v] = lds32[mrow * ldw + ((k0 + a_koff(v, half)) >> 1)];
    const unsigned int* wr =
        (const unsigned int*)(Wt + (size_t)(col0 + mrow) * K + k0 + 16 * half);
#pragma unroll
    for (int v = 0; v < 8; ++v) bf.d[v] = wr[v];
    acc = __builtin_amdgcn_wmma_f32_16x16x32_bf16(false, af.v, false, bf.v,
                                                  (short)0, acc, false, false);
  }
  float bval = bias ? bias[col0 + mrow] : 0.f;
#pragma unroll
  for (int r = 0; r < 8; ++r) {
    int row = m0 + r + 8 * half;
    C[(size_t)row * 128 + col0 + mrow] = acc[r] + bval;
  }
}

// ---------------------------------------------------------------------------
// Pair-bias kernel: per 16-pair tile builds RBF(d_ij) and outer(left_i,right_j)
// feature tiles (16x256 bf16) in LDS, then runs two WMMA chains (K=256):
//   bias_z = rbf @ emb_rbf_w + b   ;   gate = sigmoid(outer @ to_gate_w + b)
// writes prod = bias_z * gate  (f32 [NN,128])
// ---------------------------------------------------------------------------
__global__ __launch_bounds__(256)
void pair_bias_k(const float* __restrict__ coords, const float* __restrict__ left,
                 const float* __restrict__ right, const u16* __restrict__ WrbfT,
                 const u16* __restrict__ WgateT, const float* __restrict__ rbf_b,
                 const float* __restrict__ gate_b, float* __restrict__ prod) {
  __shared__ u16 rT[16 * 256];
  __shared__ u16 oT[16 * 256];
  const int tid = threadIdx.x;
  const int p0 = blockIdx.x * 16;
  {
    int r = tid >> 4;
    int kb = (tid & 15) * 16;
    int p = p0 + r, i = p / NTOK, j = p % NTOK;
    float dx = coords[i * 3 + 0] - coords[j * 3 + 0];
    float dy = coords[i * 3 + 1] - coords[j * 3 + 1];
    float dz = coords[i * 3 + 2] - coords[j * 3 + 2];
    float d = sqrtf(dx * dx + dy * dy + dz * dz + 1e-12f);
    float td = 2.f * d;
#pragma unroll
    for (int k = kb; k < kb + 16; ++k) {
      float t1 = td - (float)k;               // (d - 0.5k)/0.5
      rT[r * 256 + k] = f2bf(__expf(-t1 * t1));
      oT[r * 256 + k] = f2bf(left[i * 16 + (k >> 4)] * right[j * 16 + (k & 15)]);
    }
  }
  __syncthreads();
  const int wave = tid >> 5, lane = tid & 31;
  const int col0 = wave * 16, mrow = lane & 15, half = lane >> 4;
  f32x8 accR = {}, accG = {};
  const unsigned int* l32r = (const unsigned int*)rT;
  const unsigned int* l32o = (const unsigned int*)oT;
  for (int k0 = 0; k0 < 256; k0 += 32) {
    Frag ar, ao, br, bg;
#pragma unroll
    for (int v = 0; v < 8; ++v) {
      int idx = (mrow * 256 + k0 + a_koff(v, half)) >> 1;
      ar.d[v] = l32r[idx];
      ao.d[v] = l32o[idx];
    }
    const unsigned int* wr =
        (const unsigned int*)(WrbfT + (size_t)(col0 + mrow) * 256 + k0 + 16 * half);
    const unsigned int* wg =
        (const unsigned int*)(WgateT + (size_t)(col0 + mrow) * 256 + k0 + 16 * half);
#pragma unroll
    for (int v = 0; v < 8; ++v) { br.d[v] = wr[v]; bg.d[v] = wg[v]; }
    accR = __builtin_amdgcn_wmma_f32_16x16x32_bf16(false, ar.v, false, br.v,
                                                   (short)0, accR, false, false);
    accG = __builtin_amdgcn_wmma_f32_16x16x32_bf16(false, ao.v, false, bg.v,
                                                   (short)0, accG, false, false);
  }
#pragma unroll
  for (int r = 0; r < 8; ++r) {
    int row = r + 8 * half, col = col0 + mrow;
    float bz = accR[r] + rbf_b[col];
    float g = sigm(accG[r] + gate_b[col]);
    prod[(size_t)(p0 + row) * 128 + col] = bz * g;
  }
}

// prod[NN,128] @ to_bias_w[128,4] -> tri_bias[NN,4]
__global__ void bias_reduce_k(const float* __restrict__ prod,
                              const float* __restrict__ wb,
                              float* __restrict__ tri_bias) {
  int id = blockIdx.x * 256 + threadIdx.x;
  if (id >= NN * 4) return;
  int h = id & 3;
  size_t p = (size_t)(id >> 2);
  const float* pr = prod + p * 128;
  float acc = 0.f;
  for (int c = 0; c < 128; ++c) acc += pr[c] * wb[c * 4 + h];
  tri_bias[id] = acc;
}

// left/right projections: node[384,256] @ w[256,16] + b
__global__ void proj_lr_k(const float* __restrict__ node, const float* __restrict__ wl,
                          const float* __restrict__ bl, const float* __restrict__ wr,
                          const float* __restrict__ br, float* __restrict__ left,
                          float* __restrict__ right) {
  int id = blockIdx.x * 256 + threadIdx.x;
  if (id >= NTOK * 16) return;
  int o = id & 15, n = id >> 4;
  const float* x = node + (size_t)n * 256;
  float al = bl[o], ar = br[o];
  for (int k = 0; k < 256; ++k) {
    al += x[k] * wl[k * 16 + o];
    ar += x[k] * wr[k * 16 + o];
  }
  left[id] = al; right[id] = ar;
}

// ---------------------------------------------------------------------------
// Row layernorm over 128 channels: f32 in -> bf16 out.
// One wave per row, lane handles 4 channels; shuffle reductions, no barriers.
// ---------------------------------------------------------------------------
__global__ __launch_bounds__(256)
void ln_rows_k(const float* __restrict__ X, const float* __restrict__ g,
               const float* __restrict__ b, u16* __restrict__ Y) {
  const int wave = threadIdx.x >> 5, lane = threadIdx.x & 31;
  const size_t row = (size_t)blockIdx.x * 8 + wave;
  const float* xr = X + row * 128;
  float x0 = xr[lane], x1 = xr[lane + 32], x2 = xr[lane + 64], x3 = xr[lane + 96];
  float mean = wave_sum(x0 + x1 + x2 + x3) * (1.f / 128.f);
  float d0 = x0 - mean, d1 = x1 - mean, d2 = x2 - mean, d3 = x3 - mean;
  float var = wave_sum(d0 * d0 + d1 * d1 + d2 * d2 + d3 * d3) * (1.f / 128.f);
  float rs = rsqrtf(var + 1e-5f);
  u16* yr = Y + row * 128;
  yr[lane]      = f2bf(d0 * rs * g[lane]      + b[lane]);
  yr[lane + 32] = f2bf(d1 * rs * g[lane + 32] + b[lane + 32]);
  yr[lane + 64] = f2bf(d2 * rs * g[lane + 64] + b[lane + 64]);
  yr[lane + 96] = f2bf(d3 * rs * g[lane + 96] + b[lane + 96]);
}

// a = mask * sigmoid(rawG) * rawP, packed bf16 in [c, r1, r2] (or transposed)
__global__ void make_ab_k(const float* __restrict__ rawG, const float* __restrict__ rawP,
                          const float* __restrict__ mask, u16* __restrict__ out,
                          int transpose) {
  size_t id = (size_t)blockIdx.x * 256 + threadIdx.x;
  if (id >= (size_t)NN * 128) return;
  int c = (int)(id & 127);
  size_t p = id >> 7;
  int r1 = (int)(p / NTOK), r2 = (int)(p % NTOK);
  float v = mask[p] * sigm(rawG[id]) * rawP[id];
  size_t op = transpose ? ((size_t)r2 * NTOK + r1) : p;
  out[(size_t)c * NN + op] = f2bf(v);
}

// ---------------------------------------------------------------------------
// Triangle einsum: X[i,j,c] = sum_k P[c,i,k] * Q[c,j,k]  (per-channel GEMM)
// grid (24,24,16); wave w -> channel blockIdx.z*8 + w; 16x16 output tile.
// ---------------------------------------------------------------------------
__global__ __launch_bounds__(256)
void tri_gemm_k(const u16* __restrict__ P, const u16* __restrict__ Q,
                float* __restrict__ X) {
  const int tid = threadIdx.x;
  const int wave = tid >> 5, lane = tid & 31;
  const int c = blockIdx.z * 8 + wave;
  const int i0 = blockIdx.x * 16, j0 = blockIdx.y * 16;
  const int mrow = lane & 15, half = lane >> 4;
  const u16* Pc = P + (size_t)c * NN;
  const u16* Qc = Q + (size_t)c * NN;
  f32x8 acc = {};
  for (int k0 = 0; k0 < NTOK; k0 += 32) {
    Frag af, bf;
    const unsigned int* pr =
        (const unsigned int*)(Pc + (size_t)(i0 + mrow) * NTOK + k0);
#pragma unroll
    for (int v = 0; v < 8; ++v) af.d[v] = pr[a_koff(v, half) >> 1];
    const unsigned int* qr =
        (const unsigned int*)(Qc + (size_t)(j0 + mrow) * NTOK + k0 + 16 * half);
#pragma unroll
    for (int v = 0; v < 8; ++v) bf.d[v] = qr[v];
    acc = __builtin_amdgcn_wmma_f32_16x16x32_bf16(false, af.v, false, bf.v,
                                                  (short)0, acc, false, false);
  }
#pragma unroll
  for (int r = 0; r < 8; ++r) {
    int row = i0 + r + 8 * half, col = j0 + mrow;
    X[((size_t)row * NTOK + col) * 128 + c] = acc[r];
  }
}

// z += sigmoid(rawG) * rawP  (rawP already has +p_b from GEMM)
__global__ void add_gated_k(float* __restrict__ z, const float* __restrict__ rawG,
                            const float* __restrict__ rawP) {
  size_t id = (size_t)blockIdx.x * 256 + threadIdx.x;
  if (id >= (size_t)NN * 128) return;
  z[id] += sigm(rawG[id]) * rawP[id];
}

// ---------------------------------------------------------------------------
// kNN: 30 nearest + 2 forced linear neighbours (top-32 of -d with rules)
// ---------------------------------------------------------------------------
__global__ void knn_k(const float* __restrict__ coords, const float* __restrict__ mask,
                      int* __restrict__ idx) {
  int i = blockIdx.x * 128 + threadIdx.x;
  if (i >= NTOK) return;
  float d[NTOK];
  float xi = coords[i * 3], yi = coords[i * 3 + 1], zi = coords[i * 3 + 2];
  for (int j = 0; j < NTOK; ++j) {
    float dx = xi - coords[j * 3], dy = yi - coords[j * 3 + 1],
          dz = zi - coords[j * 3 + 2];
    float dd = sqrtf(dx * dx + dy * dy + dz * dz + 1e-12f);
    int off = (i > j) ? (i - j) : (j - i);
    if (off == 0) dd = INFV;
    else if (off <= 1) dd = 0.f;          // K_LIN/2 == 1
    dd += INFV * (mask[(size_t)i * NTOK + j] - 1.f);
    d[j] = dd;
  }
  for (int kk = 0; kk < KTOT; ++kk) {
    float best = __builtin_inff(); int bj = 0;
    for (int j = 0; j < NTOK; ++j) if (d[j] < best) { best = d[j]; bj = j; }
    idx[i * KTOT + kk] = bj;
    d[bj] = __builtin_inff();
  }
}

// gather z rows by idx (starting: z[n,j,:], ending: z[j,n,:]), LN -> bf16;
// also gather tri_bias (H) and mask bias per key. One wave per gathered row.
__global__ __launch_bounds__(256)
void gather_ln_k(const float* __restrict__ z, const int* __restrict__ idx,
                 const float* __restrict__ mask, const float* __restrict__ tri_bias,
                 const float* __restrict__ g, const float* __restrict__ b,
                 u16* __restrict__ xg, float* __restrict__ tbg,
                 float* __restrict__ maskb, int starting) {
  const int wave = threadIdx.x >> 5, lane = threadIdx.x & 31;
  const int pr = blockIdx.x * 8 + wave;      // n*32 + kk
  if (pr >= MROWS) return;
  const int n = pr >> 5;
  const int j = idx[pr];
  size_t zr = starting ? ((size_t)n * NTOK + j) : ((size_t)j * NTOK + n);
  const float* xr = z + zr * 128;
  float x0 = xr[lane], x1 = xr[lane + 32], x2 = xr[lane + 64], x3 = xr[lane + 96];
  float mean = wave_sum(x0 + x1 + x2 + x3) * (1.f / 128.f);
  float d0 = x0 - mean, d1 = x1 - mean, d2 = x2 - mean, d3 = x3 - mean;
  float var = wave_sum(d0 * d0 + d1 * d1 + d2 * d2 + d3 * d3) * (1.f / 128.f);
  float rs = rsqrtf(var + 1e-5f);
  u16* yr = xg + (size_t)pr * 128;
  yr[lane]      = f2bf(d0 * rs * g[lane]      + b[lane]);
  yr[lane + 32] = f2bf(d1 * rs * g[lane + 32] + b[lane + 32]);
  yr[lane + 64] = f2bf(d2 * rs * g[lane + 64] + b[lane + 64]);
  yr[lane + 96] = f2bf(d3 * rs * g[lane + 96] + b[lane + 96]);
  if (lane < 4) tbg[pr * 4 + lane] = tri_bias[zr * 4 + lane];
  if (lane == 4) maskb[pr] = INFV * (mask[zr] - 1.f);
}

// per-(n,h,q) softmax attention over 32 keys, C_HID=32
__global__ void attn_k(const float* __restrict__ qb, const float* __restrict__ kb,
                       const float* __restrict__ vb, const float* __restrict__ maskb,
                       const float* __restrict__ tbg, float* __restrict__ obuf) {
  int gid = blockIdx.x * 256 + threadIdx.x;
  if (gid >= NTOK * 128) return;            // N * H * 32
  int q = gid & 31, h = (gid >> 5) & 3, n = gid >> 7;
  const float* qrow = qb + (size_t)(n * KTOT + q) * 128 + h * 32;
  float s[KTOT];
  float mx = -3.4e38f;
  for (int kk = 0; kk < KTOT; ++kk) {
    const float* krow = kb + (size_t)(n * KTOT + kk) * 128 + h * 32;
    float acc = 0.f;
    for (int c = 0; c < 32; ++c) acc += qrow[c] * krow[c];
    acc = acc * 0.1767766953f + maskb[n * KTOT + kk] + tbg[(n * KTOT + kk) * 4 + h];
    s[kk] = acc;
    mx = fmaxf(mx, acc);
  }
  float den = 0.f;
  for (int kk = 0; kk < KTOT; ++kk) { s[kk] = __expf(s[kk] - mx); den += s[kk]; }
  float inv = 1.f / den;
  for (int c = 0; c < 32; ++c) {
    float o = 0.f;
    for (int kk = 0; kk < KTOT; ++kk)
      o += s[kk] * vb[(size_t)(n * KTOT + kk) * 128 + h * 32 + c];
    obuf[(size_t)(n * KTOT + q) * 128 + h * 32 + c] = o * inv;
  }
}

// og = bf16(obuf * sigmoid(graw))
__global__ void gate_pack_k(const float* __restrict__ obuf, const float* __restrict__ graw,
                            u16* __restrict__ og) {
  size_t id = (size_t)blockIdx.x * 256 + threadIdx.x;
  if (id >= (size_t)MROWS * 128) return;
  og[id] = f2bf(obuf[id] * sigm(graw[id]));
}

// scatter-add attention output back into z at gathered positions
__global__ void scatter_add_k(float* __restrict__ z, const float* __restrict__ rawO,
                              const int* __restrict__ idx, int starting) {
  size_t id = (size_t)blockIdx.x * 256 + threadIdx.x;
  if (id >= (size_t)MROWS * 128) return;
  int c = (int)(id & 127);
  int pr = (int)(id >> 7);
  int n = pr >> 5;
  int j = idx[pr];
  size_t dst = starting ? ((size_t)n * NTOK + j) : ((size_t)j * NTOK + n);
  z[dst * 128 + c] += rawO[id];
}

// ---------------------------------------------------------------------------
// Host orchestration
// ---------------------------------------------------------------------------
extern "C" void kernel_launch(void* const* d_in, const int* in_sizes, int n_in,
                              void* d_out, int out_size, void* d_ws, size_t ws_size,
                              hipStream_t stream) {
  (void)in_sizes; (void)n_in; (void)out_size; (void)ws_size;
  auto in = [&](int i) -> const float* { return (const float*)d_in[i]; };
  // input layout (setup_inputs order, params flattened in insertion order):
  // 0 node_embed 1 edge_embed 2 coords 3 edge_mask 4 contact 5 t
  // 6..16 : proj_left_w/b, proj_right_w/b, to_gate_w/b, emb_rbf_w/b, to_bias_w,
  //         ln_g, ln_b
  // 17..32: tmo_  (ag_w,ag_b,ap_w,ap_b,bg_w,bg_b,bp_w,bp_b,
  //                ln_in_g,ln_in_b,ln_out_g,ln_out_b,g_w,g_b,p_w,p_b)
  // 33..48: tmi_  (same)
  // 49..55: mhs_  (q_w,k_w,v_w,g_w,g_b,o_w,o_b)
  // 56..62: mhe_  (same)

  size_t cur = 0;
  auto alloc = [&](size_t bytes) -> size_t {
    cur = (cur + 255) & ~(size_t)255;
    size_t o = cur; cur += bytes; return o;
  };
  auto wsp = [&](size_t off) -> char* { return (char*)d_ws + off; };

  // transposed bf16 weights
  size_t wt_tri[2][6];                         // ag,ap,bg,bp,g,p (128x128)
  for (int s = 0; s < 2; ++s)
    for (int k = 0; k < 6; ++k) wt_tri[s][k] = alloc(128 * 128 * 2);
  size_t wt_mha[2][5];                         // q,k,v,g,o (128x128)
  for (int s = 0; s < 2; ++s)
    for (int k = 0; k < 5; ++k) wt_mha[s][k] = alloc(128 * 128 * 2);
  size_t wt_gate = alloc(256 * 128 * 2);
  size_t wt_rbf  = alloc(256 * 128 * 2);

  size_t leftb  = alloc(NTOK * 16 * 4);
  size_t rightb = alloc(NTOK * 16 * 4);
  size_t tri_bias = alloc((size_t)NN * 4 * 4);
  size_t prodb  = alloc((size_t)NN * 128 * 4);   // reused as xtri later
  size_t zl     = alloc((size_t)NN * 128 * 2);
  size_t raw0   = alloc((size_t)NN * 128 * 4);
  size_t raw1   = alloc((size_t)NN * 128 * 4);
  size_t rawG   = alloc((size_t)NN * 128 * 4);
  size_t abuf   = alloc((size_t)NN * 128 * 2);
  size_t bbuf   = alloc((size_t)NN * 128 * 2);
  size_t xln    = alloc((size_t)NN * 128 * 2);
  size_t idxb   = alloc(NTOK * KTOT * 4);
  size_t xg     = alloc((size_t)MROWS * 128 * 2);
  size_t tbg    = alloc(MROWS * 4 * 4);
  size_t maskb  = alloc(MROWS * 4);
  size_t qb     = alloc((size_t)MROWS * 128 * 4);
  size_t kbuf   = alloc((size_t)MROWS * 128 * 4);
  size_t vb     = alloc((size_t)MROWS * 128 * 4);
  size_t grawm  = alloc((size_t)MROWS * 128 * 4);
  size_t obuf   = alloc((size_t)MROWS * 128 * 4);
  size_t og     = alloc((size_t)MROWS * 128 * 2);
  size_t rawO   = alloc((size_t)MROWS * 128 * 4);

  float* z = (float*)d_out;
  const float* mask = in(3);

  auto T = [&](int widx, size_t dst, int K, int Nc) {
    transpose_bf16_k<<<(K * Nc + 255) / 256, 256, 0, stream>>>(in(widx), (u16*)wsp(dst), K, Nc);
  };
  auto GEMM = [&](size_t A, size_t Wt, const float* bias, size_t C, int M, int K) {
    gemm_bf16_k<<<M / 16, 256, 0, stream>>>((const u16*)wsp(A), (const u16*)wsp(Wt),
                                            bias, (float*)wsp(C), M, K);
  };

  // ---- weight conversion -------------------------------------------------
  for (int s = 0; s < 2; ++s) {
    int pb = (s == 0) ? 17 : 33;
    int wof[6] = {0, 2, 4, 6, 12, 14};     // ag_w,ap_w,bg_w,bp_w,g_w,p_w
    for (int k = 0; k < 6; ++k) T(pb + wof[k], wt_tri[s][k], 128, 128);
  }
  for (int s = 0; s < 2; ++s) {
    int pb = (s == 0) ? 49 : 56;
    int wof[5] = {0, 1, 2, 3, 5};          // q,k,v,g,o
    for (int k = 0; k < 5; ++k) T(pb + wof[k], wt_mha[s][k], 128, 128);
  }
  T(10, wt_gate, 256, 128);
  T(12, wt_rbf, 256, 128);

  // ---- z init ------------------------------------------------------------
  hipMemcpyAsync(z, d_in[1], (size_t)NN * 128 * 4, hipMemcpyDeviceToDevice, stream);

  // ---- pair bias (RBF + gate) -> tri_bias [N,N,4] ------------------------
  proj_lr_k<<<(NTOK * 16 + 255) / 256, 256, 0, stream>>>(
      in(0), in(6), in(7), in(8), in(9), (float*)wsp(leftb), (float*)wsp(rightb));
  pair_bias_k<<<NN / 16, 256, 0, stream>>>(
      in(2), (const float*)wsp(leftb), (const float*)wsp(rightb),
      (const u16*)wsp(wt_rbf), (const u16*)wsp(wt_gate), in(13), in(11),
      (float*)wsp(prodb));
  bias_reduce_k<<<(NN * 4 + 255) / 256, 256, 0, stream>>>(
      (const float*)wsp(prodb), in(14), (float*)wsp(tri_bias));

  // ---- kNN ---------------------------------------------------------------
  knn_k<<<(NTOK + 127) / 128, 128, 0, stream>>>(in(2), mask, (int*)wsp(idxb));

  // ---- triangle multiplications (outgoing, then incoming) ----------------
  for (int s = 0; s < 2; ++s) {
    int pb = (s == 0) ? 17 : 33;
    int transpose = s;                       // incoming contracts leading index
    ln_rows_k<<<NN / 8, 256, 0, stream>>>(z, in(pb + 8), in(pb + 9), (u16*)wsp(zl));
    GEMM(zl, wt_tri[s][0], in(pb + 1), raw0, NN, 128);   // ag
    GEMM(zl, wt_tri[s][1], in(pb + 3), raw1, NN, 128);   // ap
    make_ab_k<<<((size_t)NN * 128 + 255) / 256, 256, 0, stream>>>(
        (const float*)wsp(raw0), (const float*)wsp(raw1), mask, (u16*)wsp(abuf), transpose);
    GEMM(zl, wt_tri[s][2], in(pb + 5), raw0, NN, 128);   // bg
    GEMM(zl, wt_tri[s][3], in(pb + 7), raw1, NN, 128);   // bp
    make_ab_k<<<((size_t)NN * 128 + 255) / 256, 256, 0, stream>>>(
        (const float*)wsp(raw0), (const float*)wsp(raw1), mask, (u16*)wsp(bbuf), transpose);
    GEMM(zl, wt_tri[s][4], in(pb + 13), rawG, NN, 128);  // gate
    tri_gemm_k<<<dim3(NTOK / 16, NTOK / 16, 16), 256, 0, stream>>>(
        (const u16*)wsp(abuf), (const u16*)wsp(bbuf), (float*)wsp(prodb));
    ln_rows_k<<<NN / 8, 256, 0, stream>>>((const float*)wsp(prodb), in(pb + 10),
                                          in(pb + 11), (u16*)wsp(xln));
    GEMM(xln, wt_tri[s][5], in(pb + 15), raw0, NN, 128); // p
    add_gated_k<<<((size_t)NN * 128 + 255) / 256, 256, 0, stream>>>(
        z, (const float*)wsp(rawG), (const float*)wsp(raw0));
  }

  // ---- local MHA (starting, then ending) ---------------------------------
  for (int s = 0; s < 2; ++s) {
    int pb = (s == 0) ? 49 : 56;
    int starting = (s == 0) ? 1 : 0;
    gather_ln_k<<<MROWS / 8, 256, 0, stream>>>(
        z, (const int*)wsp(idxb), mask, (const float*)wsp(tri_bias),
        in(15), in(16), (u16*)wsp(xg), (float*)wsp(tbg), (float*)wsp(maskb), starting);
    GEMM(xg, wt_mha[s][0], nullptr, qb, MROWS, 128);
    GEMM(xg, wt_mha[s][1], nullptr, kbuf, MROWS, 128);
    GEMM(xg, wt_mha[s][2], nullptr, vb, MROWS, 128);
    GEMM(xg, wt_mha[s][3], in(pb + 4), grawm, MROWS, 128);
    attn_k<<<(NTOK * 128 + 255) / 256, 256, 0, stream>>>(
        (const float*)wsp(qb), (const float*)wsp(kbuf), (const float*)wsp(vb),
        (const float*)wsp(maskb), (const float*)wsp(tbg), (float*)wsp(obuf));
    gate_pack_k<<<((size_t)MROWS * 128 + 255) / 256, 256, 0, stream>>>(
        (const float*)wsp(obuf), (const float*)wsp(grawm), (u16*)wsp(og));
    GEMM(og, wt_mha[s][4], in(pb + 6), rawO, MROWS, 128);
    scatter_add_k<<<((size_t)MROWS * 128 + 255) / 256, 256, 0, stream>>>(
        z, (const float*)wsp(rawO), (const int*)wsp(idxb), starting);
  }
}
